// Attention_33535104647857
// MI455X (gfx1250) — matrix operand order repaired and checked by reading, not compiled, not run on hardware
//
#include <hip/hip_runtime.h>
#include <hip/hip_bf16.h>
#include <cstdint>

// ---------------------------------------------------------------------------
// Problem constants (from the reference): B=2, S=2048, D=1024, H=16, HD=64
// ---------------------------------------------------------------------------
constexpr int Bc  = 2;
constexpr int Sc  = 2048;
constexpr int Dc  = 1024;
constexpr int Hc  = 16;
constexpr int HDc = 64;
constexpr int HD2 = HDc / 2;
constexpr int Mrows = Bc * Sc;          // 4096 GEMM rows

typedef __attribute__((ext_vector_type(16))) __bf16 v16bf;
typedef __attribute__((ext_vector_type(8)))  __bf16 v8bf;
typedef __attribute__((ext_vector_type(4)))  __bf16 v4bf;
typedef __attribute__((ext_vector_type(8)))  float  v8f;

// ---------------------------------------------------------------------------
// WMMA helpers (CDNA5: D = A(16x32 bf16) * B(32x16 bf16) + C(16x16 f32))
// ---------------------------------------------------------------------------
static __device__ inline v8f wmma_bf16(v16bf a, v16bf b, v8f c) {
    return __builtin_amdgcn_wmma_f32_16x16x32_bf16(
        /*neg_a=*/false, a, /*neg_b=*/false, b,
        /*c_mod=*/(short)0, c, /*reuse_a=*/false, /*reuse_b=*/false);
}

// A-fragment (16x32 bf16): lane holds row M = lane%16; K =
// (lane<16?0:16)..+15 contiguous -> two contiguous 16-byte loads.
// `p` must already point at &row[kk] where kk = (lane<16?0:16).
static __device__ inline v16bf load_afrag(const __bf16* p) {
    v8bf lo = *(const v8bf*)(p);
    v8bf hi = *(const v8bf*)(p + 8);
    return __builtin_shufflevector(lo, hi,
        0,1,2,3,4,5,6,7,8,9,10,11,12,13,14,15);
}

// B-fragment (32x16 bf16): lane holds column n = lane%16,
// K = (lane<16?0:16)..+15 contiguous -> one 32-byte load.
static __device__ inline v16bf load_bfrag(const __bf16* p) {
    return *(const v16bf*)(p);
}

// Async 16B/lane global -> LDS copy (ASYNCcnt-tracked, no VGPR round trip).
static __device__ inline void async_cp16(uint32_t lds_off, const __bf16* g) {
    asm volatile("global_load_async_to_lds_b128 %0, %1, off"
                 :: "v"(lds_off), "v"((unsigned long long)(uintptr_t)g)
                 : "memory");
}

// ---------------------------------------------------------------------------
// fp32 -> bf16 conversion of x (vectorized, memory-bound, trivial vs 23.3TB/s)
// ---------------------------------------------------------------------------
__global__ void convert_x_kernel(const float* __restrict__ in,
                                 __bf16* __restrict__ out, int n4) {
    int i = blockIdx.x * blockDim.x + threadIdx.x;
    if (i >= n4) return;
    const float4 v = ((const float4*)in)[i];
    v4bf o; o.x = (__bf16)v.x; o.y = (__bf16)v.y; o.z = (__bf16)v.z; o.w = (__bf16)v.w;
    *(v4bf*)(out + (size_t)i * 4) = o;
}

// ---------------------------------------------------------------------------
// Transpose + convert weights: W[k][n] (fp32, DxD) -> Wt[n][k] (bf16).
// Transposed storage makes every WMMA B-fragment a contiguous 32B load.
// ---------------------------------------------------------------------------
__global__ void convert_wT_kernel(const float* __restrict__ W,
                                  __bf16* __restrict__ Wt) {
    __shared__ float tile[32][33];
    const int tx = threadIdx.x;          // 0..31
    const int ty = threadIdx.y;          // 0..7
    #pragma unroll
    for (int r = 0; r < 32; r += 8)
        tile[ty + r][tx] = W[(size_t)(blockIdx.y * 32 + ty + r) * Dc
                             + blockIdx.x * 32 + tx];
    __syncthreads();
    #pragma unroll
    for (int r = 0; r < 32; r += 8)
        Wt[(size_t)(blockIdx.x * 32 + ty + r) * Dc + blockIdx.y * 32 + tx]
            = (__bf16)tile[tx][ty + r];
}

// ---------------------------------------------------------------------------
// WMMA GEMM: C[M,N] = A[M,K](bf16) * Bt[N,K](bf16 transposed).
//
// Block = 256 threads (8 waves) computes a 128x64 tile. Per 32-wide K step,
// A(128x32) and B(64x32) bf16 panels are staged into LDS with
// global_load_async_to_lds_b128 (3 instructions/wave/stage = 1.5 KB/wave),
// double-buffered, and consumed via ds_load in WMMA fragment layout.
//
// Store modes:
//  0: bf16 row-major [M,N]
//  1: bf16 scatter to [B,H,S,HD]   (Q, K after projection)
//  2: bf16 scatter to [B,H,HD,S]   (V transposed, the PV B-operand layout)
//  3: fp32 row-major [M,N]         (final output projection)
// ---------------------------------------------------------------------------
static __device__ inline void stage_tiles(const __bf16* __restrict__ A,
                                          const __bf16* __restrict__ Bt,
                                          __bf16* As, __bf16* Bs,
                                          int K, int tm0, int tn0, int k0, int t) {
    {   // A chunks t and t+256 (512 x 16B chunks cover 128 rows x 32 k)
        const int r = t >> 2, ck = (t & 3) * 8;
        async_cp16((uint32_t)(uintptr_t)(As + r * 32 + ck),
                   A + (size_t)(tm0 + r) * K + k0 + ck);
    }
    {
        const int c = t + 256;
        const int r = c >> 2, ck = (c & 3) * 8;
        async_cp16((uint32_t)(uintptr_t)(As + r * 32 + ck),
                   A + (size_t)(tm0 + r) * K + k0 + ck);
    }
    {   // B chunk t (256 x 16B chunks cover 64 cols x 32 k)
        const int r = t >> 2, ck = (t & 3) * 8;
        async_cp16((uint32_t)(uintptr_t)(Bs + r * 32 + ck),
                   Bt + (size_t)(tn0 + r) * K + k0 + ck);
    }
}

__global__ __launch_bounds__(256)
void gemm_wmma_kernel(const __bf16* __restrict__ A,
                      const __bf16* __restrict__ Bt,
                      __bf16* __restrict__ outb,
                      float* __restrict__ outf,
                      int M, int N, int K, int mode) {
    __shared__ __bf16 As[2][128 * 32];   // 2 x 8 KB
    __shared__ __bf16 Bs[2][64 * 32];    // 2 x 4 KB

    const int t      = threadIdx.x;
    const int lane   = t & 31;
    const int lane15 = lane & 15;
    const int kk     = (lane < 16) ? 0 : 16;   // A-fragment chunk base
    const int kb16   = (lane < 16) ? 0 : 16;   // B-fragment K base
    const int w      = t >> 5;                 // wave = M subtile 0..7
    const int ntn    = N / 64;
    const int tm0    = (blockIdx.x / ntn) * 128;
    const int tn0    = (blockIdx.x % ntn) * 64;

    v8f acc[4] = {};
    const int nst = K / 32;

    stage_tiles(A, Bt, As[0], Bs[0], K, tm0, tn0, 0, t);

    for (int s = 0; s < nst; ++s) {
        const int cur = s & 1;
        if (s + 1 < nst) {
            stage_tiles(A, Bt, As[cur ^ 1], Bs[cur ^ 1], K, tm0, tn0,
                        (s + 1) * 32, t);
            // >3 outstanding would include the current stage; async loads
            // complete in order, so <=3 left  ==>  current panel has landed.
            asm volatile("s_wait_asynccnt 3" ::: "memory");
        } else {
            asm volatile("s_wait_asynccnt 0" ::: "memory");
        }
        __syncthreads();

        const v16bf af = load_afrag(&As[cur][(w * 16 + lane15) * 32 + kk]);
        #pragma unroll
        for (int j = 0; j < 4; ++j) {
            const v16bf bf = load_bfrag(&Bs[cur][(j * 16 + lane15) * 32 + kb16]);
            acc[j] = wmma_bf16(af, bf, acc[j]);
        }
        __syncthreads();   // all waves done reading before buffer is re-filled
    }

    const int mbase = tm0 + w * 16 + ((lane >> 4) << 3);
    #pragma unroll
    for (int j = 0; j < 4; ++j) {
        const int col = tn0 + j * 16 + lane15;
        #pragma unroll
        for (int i = 0; i < 8; ++i) {
            const int row = mbase + i;
            const float v = acc[j][i];
            if (mode == 0) {
                outb[(size_t)row * N + col] = (__bf16)v;
            } else if (mode == 1) {
                const int bb = row / Sc, s2 = row % Sc;
                const int hh = col / HDc, hd = col % HDc;
                outb[(((size_t)(bb * Hc + hh)) * Sc + s2) * HDc + hd] = (__bf16)v;
            } else if (mode == 2) {
                const int bb = row / Sc, s2 = row % Sc;
                const int hh = col / HDc, hd = col % HDc;
                outb[(((size_t)(bb * Hc + hh)) * HDc + hd) * Sc + s2] = (__bf16)v;
            } else {
                outf[(size_t)row * N + col] = v;
            }
        }
    }
}

// ---------------------------------------------------------------------------
// RoPE on Q and K in [B,H,S,HD] bf16 layout (pairs are adjacent in HD).
// ---------------------------------------------------------------------------
__global__ void rope_kernel(__bf16* __restrict__ Q, __bf16* __restrict__ Kt,
                            const float* __restrict__ fcos,
                            const float* __restrict__ fsin) {
    const int idx = blockIdx.x * blockDim.x + threadIdx.x;
    const int d2  = idx & (HD2 - 1);
    const int s   = (idx >> 5) & (Sc - 1);
    const int bh  = idx >> 16;                 // 32*2048 = 2^16
    if (bh >= Bc * Hc) return;
    const size_t base = ((size_t)bh * Sc + s) * HDc + 2 * d2;
    const float c  = fcos[s * HD2 + d2];
    const float sn = fsin[s * HD2 + d2];
    float qr = (float)Q[base], qi = (float)Q[base + 1];
    Q[base]     = (__bf16)(qr * c - qi * sn);
    Q[base + 1] = (__bf16)(qr * sn + qi * c);
    float kr = (float)Kt[base], ki = (float)Kt[base + 1];
    Kt[base]     = (__bf16)(kr * c - ki * sn);
    Kt[base + 1] = (__bf16)(kr * sn + ki * c);
}

// ---------------------------------------------------------------------------
// Flash attention (causal). One wave = one 16-row query tile of one (b,h).
// Key blocks of 32; online softmax; P re-swizzled via per-wave LDS.
// ---------------------------------------------------------------------------
static __device__ inline float rowmax16(float v) {
    #pragma unroll
    for (int off = 8; off >= 1; off >>= 1)
        v = fmaxf(v, __shfl_xor(v, off, 32));
    return v;
}
static __device__ inline float rowsum16(float v) {
    #pragma unroll
    for (int off = 8; off >= 1; off >>= 1)
        v += __shfl_xor(v, off, 32);
    return v;
}

__global__ void flash_attn_kernel(const __bf16* __restrict__ Q,
                                  const __bf16* __restrict__ Kt,
                                  const __bf16* __restrict__ Vt,
                                  __bf16* __restrict__ ctx) {
    __shared__ __bf16 plds[4][16 * 32];        // per-wave P tile (16x32 bf16)

    const int lane   = threadIdx.x & 31;
    const int lane15 = lane & 15;
    const int kk     = (lane < 16) ? 0 : 16;
    const int kb16   = (lane < 16) ? 0 : 16;
    const int wv     = threadIdx.x >> 5;
    const int gw     = blockIdx.x * 4 + wv;
    const int qt = gw & 127;                   // S/16 = 128 query tiles
    const int h  = (gw >> 7) & (Hc - 1);
    const int b  = gw >> 11;
    const int q0 = qt * 16;

    const __bf16* Qh = Q  + ((size_t)(b * Hc + h)) * Sc * HDc;
    const __bf16* Kh = Kt + ((size_t)(b * Hc + h)) * Sc * HDc;
    const __bf16* Vh = Vt + ((size_t)(b * Hc + h)) * HDc * Sc;

    // Q A-fragments for HD = 64 (two K-steps of 32), rows q0 + lane%16.
    const int mq = q0 + lane15;
    const v16bf qa0 = load_afrag(Qh + (size_t)mq * HDc + 0  + kk);
    const v16bf qa1 = load_afrag(Qh + (size_t)mq * HDc + 32 + kk);

    const float scale = 0.125f;                // 1/sqrt(HD)
    const int mlocal = (lane >> 4) << 3;       // this lane's row base 0 or 8
    float rmax[8], rsum[8];
    v8f o[4] = {};
    #pragma unroll
    for (int i = 0; i < 8; ++i) { rmax[i] = -3.0e38f; rsum[i] = 0.f; }

    const int kmax = q0 + 15;                  // causal: last key needed
    for (int key0 = 0; key0 <= kmax; key0 += 32) {
        const bool do1 = (key0 + 16) <= kmax;

        // ---- scores: S = Q * K^T (K tensor rows are the transposed-B cols)
        v8f s0 = {}, s1 = {};
        {
            const __bf16* kp = Kh + (size_t)(key0 + lane15) * HDc;
            s0 = wmma_bf16(qa0, load_bfrag(kp + kb16),      s0);
            s0 = wmma_bf16(qa1, load_bfrag(kp + 32 + kb16), s0);
        }
        if (do1) {
            const __bf16* kp = Kh + (size_t)(key0 + 16 + lane15) * HDc;
            s1 = wmma_bf16(qa0, load_bfrag(kp + kb16),      s1);
            s1 = wmma_bf16(qa1, load_bfrag(kp + 32 + kb16), s1);
        }

        // ---- scale + causal mask (lane owns col n, rows mlocal..mlocal+7)
        const int key_a = key0 + lane15;
        const int key_b = key0 + 16 + lane15;
        #pragma unroll
        for (int i = 0; i < 8; ++i) {
            const int qrow = q0 + mlocal + i;
            float a = s0[i] * scale;
            if (key_a > qrow) a = -3.0e38f;
            s0[i] = a;
            float bb2 = do1 ? (s1[i] * scale) : -3.0e38f;
            if (key_b > qrow) bb2 = -3.0e38f;
            s1[i] = bb2;
        }

        // ---- online softmax update
        float p0[8], p1[8], alpha[8];
        #pragma unroll
        for (int i = 0; i < 8; ++i) {
            const float bm = rowmax16(fmaxf(s0[i], s1[i]));
            const float mn = fmaxf(rmax[i], bm);
            p0[i] = __expf(s0[i] - mn);
            p1[i] = __expf(s1[i] - mn);
            const float bs = rowsum16(p0[i] + p1[i]);
            alpha[i] = __expf(rmax[i] - mn);
            rmax[i] = mn;
            rsum[i] = rsum[i] * alpha[i] + bs;
        }
        #pragma unroll
        for (int j = 0; j < 4; ++j)
            #pragma unroll
            for (int i = 0; i < 8; ++i)
                o[j][i] *= alpha[i];

        // ---- C-layout P -> bf16 A-layout, via per-wave LDS tile
        __bf16* pl = plds[wv];
        #pragma unroll
        for (int i = 0; i < 8; ++i) {
            const int r = mlocal + i;
            pl[r * 32 + lane15]      = (__bf16)p0[i];
            pl[r * 32 + 16 + lane15] = (__bf16)p1[i];
        }
        asm volatile("s_wait_dscnt 0" ::: "memory");   // wave-local LDS RAW
        const v16bf pa = load_afrag(pl + lane15 * 32 + kk);

        // ---- O += P * V   (Vt rows: hd-major, keys contiguous = B layout)
        #pragma unroll
        for (int j = 0; j < 4; ++j) {
            const __bf16* vp = Vh + (size_t)(j * 16 + lane15) * Sc + key0 + kb16;
            o[j] = wmma_bf16(pa, load_bfrag(vp), o[j]);
        }
    }

    // ---- normalize and store context as [B*S, D] bf16 (row-major for proj)
    float inv[8];
    #pragma unroll
    for (int i = 0; i < 8; ++i) inv[i] = 1.0f / rsum[i];
    #pragma unroll
    for (int j = 0; j < 4; ++j) {
        const int col = h * HDc + j * 16 + lane15;
        #pragma unroll
        for (int i = 0; i < 8; ++i) {
            const int row = q0 + mlocal + i;
            ctx[((size_t)b * Sc + row) * Dc + col] = (__bf16)(o[j][i] * inv[i]);
        }
    }
}

// ---------------------------------------------------------------------------
// Host launcher
// ---------------------------------------------------------------------------
extern "C" void kernel_launch(void* const* d_in, const int* in_sizes, int n_in,
                              void* d_out, int out_size, void* d_ws, size_t ws_size,
                              hipStream_t stream) {
    const float* x    = (const float*)d_in[0];
    const float* wq   = (const float*)d_in[1];
    const float* wk   = (const float*)d_in[2];
    const float* wv   = (const float*)d_in[3];
    const float* wo   = (const float*)d_in[4];
    const float* fcos = (const float*)d_in[5];
    const float* fsin = (const float*)d_in[6];
    // d_in[7] (mask) replaced by analytic causal mask; d_in[8] (start_pos)=0.
    float* out = (float*)d_out;

    // ---- workspace layout (bf16), ~48 MB total
    const size_t NX = (size_t)Mrows * Dc;        // 4,194,304
    const size_t NW = (size_t)Dc * Dc;           // 1,048,576
    const size_t NQ = (size_t)Bc * Hc * Sc * HDc;// 4,194,304
    __bf16* Xb  = (__bf16*)d_ws;
    __bf16* Wqt = Xb  + NX;
    __bf16* Wkt = Wqt + NW;
    __bf16* Wvt = Wkt + NW;
    __bf16* Wot = Wvt + NW;
    __bf16* Qh  = Wot + NW;
    __bf16* Kh  = Qh  + NQ;
    __bf16* Vt  = Kh  + NQ;
    __bf16* ctx = Vt  + NQ;

    // 1) convert activations / weights to bf16 (weights transposed)
    convert_x_kernel<<<(NX / 4 + 255) / 256, 256, 0, stream>>>(x, Xb, (int)(NX / 4));
    convert_wT_kernel<<<dim3(32, 32), dim3(32, 8), 0, stream>>>(wq, Wqt);
    convert_wT_kernel<<<dim3(32, 32), dim3(32, 8), 0, stream>>>(wk, Wkt);
    convert_wT_kernel<<<dim3(32, 32), dim3(32, 8), 0, stream>>>(wv, Wvt);
    convert_wT_kernel<<<dim3(32, 32), dim3(32, 8), 0, stream>>>(wo, Wot);

    // 2) QKV projections (WMMA), scattering into attention-friendly layouts
    const int gemm_blocks = (Mrows / 128) * (Dc / 64);      // 512
    gemm_wmma_kernel<<<gemm_blocks, 256, 0, stream>>>(Xb, Wqt, Qh, nullptr,
                                                      Mrows, Dc, Dc, 1);
    gemm_wmma_kernel<<<gemm_blocks, 256, 0, stream>>>(Xb, Wkt, Kh, nullptr,
                                                      Mrows, Dc, Dc, 1);
    gemm_wmma_kernel<<<gemm_blocks, 256, 0, stream>>>(Xb, Wvt, Vt, nullptr,
                                                      Mrows, Dc, Dc, 2);

    // 3) RoPE on Q and K
    const int rope_threads = Bc * Hc * Sc * HD2;            // 2,097,152
    rope_kernel<<<rope_threads / 256, 256, 0, stream>>>(Qh, Kh, fcos, fsin);

    // 4) causal flash attention
    const int fa_blocks = Bc * Hc * (Sc / 16) / 4;          // 1024
    flash_attn_kernel<<<fa_blocks, 128, 0, stream>>>(Qh, Kh, Vt, ctx);

    // 5) output projection -> fp32 d_out
    gemm_wmma_kernel<<<gemm_blocks, 256, 0, stream>>>(ctx, Wot, nullptr, out,
                                                      Mrows, Dc, Dc, 3);
}
